// GatedAttn_17557826306464
// MI455X (gfx1250) — matrix operand order repaired and checked by reading, not compiled, not run on hardware
//
#include <hip/hip_runtime.h>
#include <hip/hip_bf16.h>
#include <math.h>

// Problem sizes (match reference)
#define B_   2
#define S_   2048
#define D_   1024
#define H_   16
#define HD_  64
#define NK_  21
#define NTOK (B_ * S_)          // 4096
#define N3   (3 * D_)           // 3072
#define N2   (2 * D_)           // 2048
#define RLEN (2 * S_ - 1)       // 4095

typedef __bf16 bf16;
typedef __attribute__((ext_vector_type(8)))  __bf16 bf16x8;
typedef __attribute__((ext_vector_type(16))) __bf16 bf16x16;
typedef __attribute__((ext_vector_type(8)))  float  f32x8;
typedef __attribute__((ext_vector_type(4)))  int    i32x4;

#define AS1 __attribute__((address_space(1)))
#define AS3 __attribute__((address_space(3)))

#if __has_builtin(__builtin_amdgcn_global_load_async_to_lds_b128)
#define HAVE_ASYNC_LDS 1
#else
#define HAVE_ASYNC_LDS 0
#endif

// 16-byte global -> LDS copy (async path; verified signature:
// (v4i addrspace(1)* src, v4i addrspace(3)* dst, imm offset, imm cpol)).
static __device__ inline void cp16(const bf16* __restrict__ g, bf16* l) {
#if HAVE_ASYNC_LDS
  __builtin_amdgcn_global_load_async_to_lds_b128(
      (AS1 i32x4*)g, (AS3 i32x4*)l, 0, 0);
#else
  *(bf16x8*)l = *(const bf16x8*)g;
#endif
}

// Wait for this wave's async copies, then make LDS visible block-wide.
static __device__ inline void stage_join() {
#if HAVE_ASYNC_LDS
#if __has_builtin(__builtin_amdgcn_s_wait_asynccnt)
  __builtin_amdgcn_s_wait_asynccnt(0);
#else
  asm volatile("s_wait_asynccnt 0" ::: "memory");
#endif
#endif
  __syncthreads();
}

static __device__ inline bf16x16 frag_cat(bf16x8 lo, bf16x8 hi) {
  return __builtin_shufflevector(lo, hi, 0,1,2,3,4,5,6,7,8,9,10,11,12,13,14,15);
}

static __device__ inline f32x8 wmma_bf16(bf16x16 a, bf16x16 b, f32x8 c) {
  // 8-arg form: (neg_a, A, neg_b, B, c_mod, C, reuse_a, reuse_b)
  return __builtin_amdgcn_wmma_f32_16x16x32_bf16(false, a, false, b, (short)0, c,
                                                 false, false);
}

// A-fragment (16x32 bf16, M x K), row-major source, row stride ld.
// Lane L: row M = L&15; K = (L>>4)*8 + 0..7 and 16 + (L>>4)*8 + 0..7.
static __device__ inline bf16x16 load_afrag(const bf16* base, int ld,
                                            int row0, int k0, int lane) {
  const bf16* p = base + (size_t)(row0 + (lane & 15)) * ld + k0 + ((lane >> 4) << 3);
  bf16x8 lo = *(const bf16x8*)p;
  bf16x8 hi = *(const bf16x8*)(p + 16);
  return frag_cat(lo, hi);
}

// B-fragment (32x16 bf16, K x N) via transposed storage Bt (N x K row-major):
// lane L holds column N = L&15, K = (L>>4)*8 + 0..7 and 16 + (L>>4)*8 + 0..7.
static __device__ inline bf16x16 load_bfrag(const bf16* bt, int ld,
                                            int col0, int k0, int lane) {
  const bf16* p = bt + (size_t)(col0 + (lane & 15)) * ld + k0 + ((lane >> 4) << 3);
  bf16x8 lo = *(const bf16x8*)p;
  bf16x8 hi = *(const bf16x8*)(p + 16);
  return frag_cat(lo, hi);
}

// ---------------------------------------------------------------- conversions
__global__ void cvt_bf16_kernel(const float* __restrict__ in, bf16* __restrict__ out,
                                int n) {
  int i = blockIdx.x * blockDim.x + threadIdx.x;
  if (i < n) out[i] = (bf16)in[i];
}

// in: rows x cols fp32 (row-major)  ->  out: cols x rows bf16 (row-major)
__global__ void transpose_bf16_kernel(const float* __restrict__ in,
                                      bf16* __restrict__ out, int rows, int cols) {
  __shared__ float tile[32][33];
  int bx = blockIdx.x * 32, by = blockIdx.y * 32;
  int tx = threadIdx.x, ty = threadIdx.y;
#pragma unroll
  for (int i = 0; i < 4; ++i)
    tile[ty + i * 8][tx] = in[(size_t)(by + ty + i * 8) * cols + bx + tx];
  __syncthreads();
#pragma unroll
  for (int i = 0; i < 4; ++i)
    out[(size_t)(bx + ty + i * 8) * rows + by + tx] = (bf16)tile[tx][ty + i * 8];
}

// ---------------------------------------------------------------- TISA bias
__global__ void tisa_bias_kernel(const float* __restrict__ amp,
                                 const float* __restrict__ shp,
                                 const float* __restrict__ off,
                                 float* __restrict__ bias) {
  int i = blockIdx.x * blockDim.x + threadIdx.x;
  if (i >= H_ * RLEN) return;
  int h = i / RLEN, r = i % RLEN;
  float rel = (float)(r - (S_ - 1));
  float s = 0.f;
#pragma unroll
  for (int k = 0; k < NK_; ++k) {
    float d = rel - off[h * NK_ + k];
    s += amp[h * NK_ + k] * __expf(-fabsf(shp[h * NK_ + k]) * d * d);
  }
  bias[i] = s;
}

// ------------------------------------------------------- QKV projection GEMM
// proj = X(4096x1024) @ Win(1024x3072), split [k|v|q]. Block tile 128x128,
// K-panel 64, double-buffered async LDS staging; 8 waves x (32x64) register
// tile with individually named accumulators (16 WMMA per barrier).
__global__ void __launch_bounds__(256) gemm_qkv_kernel(
    const bf16* __restrict__ X, const bf16* __restrict__ WinT,
    bf16* __restrict__ Q, bf16* __restrict__ Kc, bf16* __restrict__ Vt) {
  __shared__ __align__(16) bf16 As[2][128][64];
  __shared__ __align__(16) bf16 Bs[2][128][64];
  int tid = threadIdx.x, lane = tid & 31, wave = tid >> 5;
  int kh = lane >> 4, n = lane & 15;
  const int NB = N3 / 128;                    // 24 column blocks
  int m0b = (blockIdx.x / NB) * 128;
  int n0b = (blockIdx.x % NB) * 128;
  int wm = wave >> 1, wn = wave & 1;          // 4x2 wave grid

  auto stage = [&](int buf, int k0) {
#pragma unroll
    for (int i = 0; i < 4; ++i) {             // 1024 16B chunks per matrix
      int c = tid * 4 + i, r = c >> 3, o = (c & 7) * 8;
      cp16(X + (size_t)(m0b + r) * D_ + k0 + o, &As[buf][r][o]);
      cp16(WinT + (size_t)(n0b + r) * D_ + k0 + o, &Bs[buf][r][o]);
    }
  };

  f32x8 c00 = {}, c01 = {}, c02 = {}, c03 = {};
  f32x8 c10 = {}, c11 = {}, c12 = {}, c13 = {};
  stage(0, 0);
  stage_join();
  for (int ks = 0; ks < D_ / 64; ++ks) {
    int buf = ks & 1;
    if (ks + 1 < D_ / 64) stage(buf ^ 1, (ks + 1) * 64);
#pragma unroll
    for (int kk = 0; kk < 64; kk += 32) {
      bf16x16 a0 = load_afrag(&As[buf][0][0], 64, wm * 32,      kk, lane);
      bf16x16 a1 = load_afrag(&As[buf][0][0], 64, wm * 32 + 16, kk, lane);
      bf16x16 b0 = load_bfrag(&Bs[buf][0][0], 64, wn * 64 +  0, kk, lane);
      bf16x16 b1 = load_bfrag(&Bs[buf][0][0], 64, wn * 64 + 16, kk, lane);
      bf16x16 b2 = load_bfrag(&Bs[buf][0][0], 64, wn * 64 + 32, kk, lane);
      bf16x16 b3 = load_bfrag(&Bs[buf][0][0], 64, wn * 64 + 48, kk, lane);
      c00 = wmma_bf16(a0, b0, c00);  c01 = wmma_bf16(a0, b1, c01);
      c02 = wmma_bf16(a0, b2, c02);  c03 = wmma_bf16(a0, b3, c03);
      c10 = wmma_bf16(a1, b0, c10);  c11 = wmma_bf16(a1, b1, c11);
      c12 = wmma_bf16(a1, b2, c12);  c13 = wmma_bf16(a1, b3, c13);
    }
    stage_join();
  }

  int n0 = n0b + wn * 64;
  int sect = n0 >> 10;                        // 0:k 1:v 2:q (reference order)
  int h = (n0 & (D_ - 1)) >> 6;               // 64-aligned tile -> single head
  auto store_tile = [&](f32x8 v, int m0, int t) {
    int hd = t * 16 + n;
#pragma unroll
    for (int r = 0; r < 8; ++r) {
      int tok = m0 + r + (kh << 3);
      int bb = tok >> 11, s = tok & (S_ - 1);
      bf16 w = (bf16)v[r];
      size_t bh = (size_t)(bb * H_ + h);
      if (sect == 0)      Kc[(bh * S_ + s) * HD_ + hd] = w;
      else if (sect == 2) Q [(bh * S_ + s) * HD_ + hd] = w;
      else                Vt[(bh * HD_ + hd) * (size_t)S_ + s] = w;
    }
  };
  int mA = m0b + wm * 32, mB = mA + 16;
  store_tile(c00, mA, 0); store_tile(c01, mA, 1);
  store_tile(c02, mA, 2); store_tile(c03, mA, 3);
  store_tile(c10, mB, 0); store_tile(c11, mB, 1);
  store_tile(c12, mB, 2); store_tile(c13, mB, 3);
}

// ------------------------------------------------------- flash attention
// One block = one (b,h) and 8 query tiles (one per wave); K/V tiles (32 keys)
// are staged once per block in double-buffered LDS and shared by all waves.
__global__ void __launch_bounds__(256) attn_kernel(
    const bf16* __restrict__ Q, const bf16* __restrict__ Kc,
    const bf16* __restrict__ Vt, const float* __restrict__ bias,
    bf16* __restrict__ O) {
  __shared__ __align__(16) bf16 Ks[2][32][64];     // [key][hd]
  __shared__ __align__(16) bf16 Vs[2][64][32];     // [hd][key-local]
  __shared__ __align__(16) bf16 Plds[8][16][32];   // per-wave P staging
  int tid = threadIdx.x, lane = tid & 31, wave = tid >> 5;
  int kh = lane >> 4, n = lane & 15;
  int gid = blockIdx.x;                      // 512 = B*H*(S/16/8)
  int qg = gid & 15;
  int h  = (gid >> 4) & (H_ - 1);
  int b  = gid >> 8;
  int q0 = (qg * 8 + wave) * 16;

  const bf16* Qb = Q  + (size_t)(b * H_ + h) * S_ * HD_;
  const bf16* Kb = Kc + (size_t)(b * H_ + h) * S_ * HD_;
  const bf16* Vb = Vt + (size_t)(b * H_ + h) * HD_ * S_;
  const float* br = bias + (size_t)h * RLEN;

  auto stage = [&](int buf, int j0) {
    int kr = tid >> 3, ko = (tid & 7) * 8;   // 32x64: 256 x 16B chunks
    cp16(Kb + (size_t)(j0 + kr) * HD_ + ko, &Ks[buf][kr][ko]);
    int vr = tid >> 2, vo = (tid & 3) * 8;   // 64x32: 256 x 16B chunks
    cp16(Vb + (size_t)vr * S_ + j0 + vo, &Vs[buf][vr][vo]);
  };

  bf16x16 qa0 = load_afrag(Qb, HD_, q0, 0, lane);    // Q K-dim [0,32)
  bf16x16 qa1 = load_afrag(Qb, HD_, q0, 32, lane);   // Q K-dim [32,64)

  f32x8 o0 = {}, o1 = {}, o2 = {}, o3 = {};
  float rmax[8], rsum[8];
#pragma unroll
  for (int r = 0; r < 8; ++r) { rmax[r] = -1e30f; rsum[r] = 0.f; }
  const float scale = 0.125f;                // 1/sqrt(HD)

  stage(0, 0);
  stage_join();
  for (int js = 0; js < S_ / 32; ++js) {
    int buf = js & 1, j0 = js * 32;
    if (js + 1 < S_ / 32) stage(buf ^ 1, j0 + 32);

    f32x8 s0 = {}, s1 = {};
    s0 = wmma_bf16(qa0, load_bfrag(&Ks[buf][0][0], 64, 0,  0,  lane), s0);
    s0 = wmma_bf16(qa1, load_bfrag(&Ks[buf][0][0], 64, 0,  32, lane), s0);
    s1 = wmma_bf16(qa0, load_bfrag(&Ks[buf][0][0], 64, 16, 0,  lane), s1);
    s1 = wmma_bf16(qa1, load_bfrag(&Ks[buf][0][0], 64, 16, 32, lane), s1);

    float alpha[8];
#pragma unroll
    for (int r = 0; r < 8; ++r) {
      int qrow = q0 + r + (kh << 3);
      int base = (S_ - 1) - qrow + j0 + n;   // rel-pos index for key j0+n
      float v0 = s0[r] * scale + br[base];
      float v1 = s1[r] * scale + br[base + 16];
      float mx = fmaxf(v0, v1);              // row max over 16-lane half
      mx = fmaxf(mx, __shfl_xor(mx, 1, 32));
      mx = fmaxf(mx, __shfl_xor(mx, 2, 32));
      mx = fmaxf(mx, __shfl_xor(mx, 4, 32));
      mx = fmaxf(mx, __shfl_xor(mx, 8, 32));
      float mnew = fmaxf(rmax[r], mx);
      alpha[r] = __expf(rmax[r] - mnew);
      rmax[r] = mnew;
      float p0 = __expf(v0 - mnew);
      float p1 = __expf(v1 - mnew);
      s0[r] = p0; s1[r] = p1;
      float ps = p0 + p1;
      ps += __shfl_xor(ps, 1, 32);
      ps += __shfl_xor(ps, 2, 32);
      ps += __shfl_xor(ps, 4, 32);
      ps += __shfl_xor(ps, 8, 32);
      rsum[r] = rsum[r] * alpha[r] + ps;
    }
#pragma unroll
    for (int r = 0; r < 8; ++r) {
      o0[r] *= alpha[r]; o1[r] *= alpha[r]; o2[r] *= alpha[r]; o3[r] *= alpha[r];
    }
    // Restage P: C-layout (row = r + 8*kh, col = n) -> A-fragment layout.
#pragma unroll
    for (int r = 0; r < 8; ++r) {
      int row = r + (kh << 3);
      Plds[wave][row][n]      = (bf16)s0[r];
      Plds[wave][row][16 + n] = (bf16)s1[r];
    }
    asm volatile("s_wait_dscnt 0" ::: "memory");
    bf16x16 pa = load_afrag(&Plds[wave][0][0], 32, 0, 0, lane);
    o0 = wmma_bf16(pa, load_bfrag(&Vs[buf][0][0], 32, 0,  0, lane), o0);
    o1 = wmma_bf16(pa, load_bfrag(&Vs[buf][0][0], 32, 16, 0, lane), o1);
    o2 = wmma_bf16(pa, load_bfrag(&Vs[buf][0][0], 32, 32, 0, lane), o2);
    o3 = wmma_bf16(pa, load_bfrag(&Vs[buf][0][0], 32, 48, 0, lane), o3);
    stage_join();
  }

#pragma unroll
  for (int r = 0; r < 8; ++r) {
    float inv = 1.f / rsum[r];
    int tok = b * S_ + q0 + r + (kh << 3);
    bf16* orow = O + (size_t)tok * D_ + h * HD_;
    orow[ 0 + n] = (bf16)(o0[r] * inv);
    orow[16 + n] = (bf16)(o1[r] * inv);
    orow[32 + n] = (bf16)(o2[r] * inv);
    orow[48 + n] = (bf16)(o3[r] * inv);
  }
}

// ------------------------------------------------------- gated output GEMM
// gated = attn(4096x1024) @ w_gate(1024x2048) + b; out = a * sigmoid(g).
// Block tile: 128 rows x 64 cols of both halves (B panel = 64 a-rows + 64
// g-rows of WgT); K-panel 64; named accumulators (16 WMMA per barrier).
__global__ void __launch_bounds__(256) gate_kernel(
    const bf16* __restrict__ A, const bf16* __restrict__ WgT,
    const float* __restrict__ bg, float* __restrict__ out) {
  __shared__ __align__(16) bf16 As[2][128][64];
  __shared__ __align__(16) bf16 Bs[2][128][64];
  int tid = threadIdx.x, lane = tid & 31, wave = tid >> 5;
  int kh = lane >> 4, n = lane & 15;
  const int CB = D_ / 64;                    // 16 column blocks
  int m0b = (blockIdx.x / CB) * 128;
  int c0b = (blockIdx.x % CB) * 64;
  int wm = wave >> 1, wn = wave & 1;

  auto stage = [&](int buf, int k0) {
#pragma unroll
    for (int i = 0; i < 4; ++i) {
      int c = tid * 4 + i, r = c >> 3, o = (c & 7) * 8;
      cp16(A + (size_t)(m0b + r) * D_ + k0 + o, &As[buf][r][o]);
      int wrow = (r < 64) ? (c0b + r) : (D_ + c0b + r - 64);  // a-half | g-half
      cp16(WgT + (size_t)wrow * D_ + k0 + o, &Bs[buf][r][o]);
    }
  };

  f32x8 a00 = {}, a01 = {}, a10 = {}, a11 = {};
  f32x8 g00 = {}, g01 = {}, g10 = {}, g11 = {};
  stage(0, 0);
  stage_join();
  for (int ks = 0; ks < D_ / 64; ++ks) {
    int buf = ks & 1;
    if (ks + 1 < D_ / 64) stage(buf ^ 1, (ks + 1) * 64);
#pragma unroll
    for (int kk = 0; kk < 64; kk += 32) {
      bf16x16 f0 = load_afrag(&As[buf][0][0], 64, wm * 32,      kk, lane);
      bf16x16 f1 = load_afrag(&As[buf][0][0], 64, wm * 32 + 16, kk, lane);
      bf16x16 bA0 = load_bfrag(&Bs[buf][0][0], 64, wn * 32,           kk, lane);
      bf16x16 bA1 = load_bfrag(&Bs[buf][0][0], 64, wn * 32 + 16,      kk, lane);
      bf16x16 bG0 = load_bfrag(&Bs[buf][0][0], 64, 64 + wn * 32,      kk, lane);
      bf16x16 bG1 = load_bfrag(&Bs[buf][0][0], 64, 64 + wn * 32 + 16, kk, lane);
      a00 = wmma_bf16(f0, bA0, a00);  a01 = wmma_bf16(f0, bA1, a01);
      a10 = wmma_bf16(f1, bA0, a10);  a11 = wmma_bf16(f1, bA1, a11);
      g00 = wmma_bf16(f0, bG0, g00);  g01 = wmma_bf16(f0, bG1, g01);
      g10 = wmma_bf16(f1, bG0, g10);  g11 = wmma_bf16(f1, bG1, g11);
    }
    stage_join();
  }

  auto emit = [&](f32x8 av8, f32x8 gv8, int m0, int u) {
    int c = c0b + wn * 32 + u * 16;
    float ba = bg[c + n];
    float bb = bg[D_ + c + n];
#pragma unroll
    for (int r = 0; r < 8; ++r) {
      float av = av8[r] + ba;
      float gv = gv8[r] + bb;
      float sig = 1.f / (1.f + __expf(-gv));
      int tok = m0 + r + (kh << 3);
      out[(size_t)tok * D_ + c + n] = av * sig;
    }
  };
  int mA = m0b + wm * 32, mB = mA + 16;
  emit(a00, g00, mA, 0); emit(a01, g01, mA, 1);
  emit(a10, g10, mB, 0); emit(a11, g11, mB, 1);
}

// ---------------------------------------------------------------- launcher
extern "C" void kernel_launch(void* const* d_in, const int* in_sizes, int n_in,
                              void* d_out, int out_size, void* d_ws, size_t ws_size,
                              hipStream_t stream) {
  const float* x      = (const float*)d_in[0];
  const float* w_in   = (const float*)d_in[1];
  const float* w_gate = (const float*)d_in[2];
  const float* b_gate = (const float*)d_in[3];
  const float* amp    = (const float*)d_in[4];
  const float* shp    = (const float*)d_in[5];
  const float* off    = (const float*)d_in[6];
  float* out = (float*)d_out;
  (void)in_sizes; (void)n_in; (void)out_size; (void)ws_size;

  char* ws = (char*)d_ws;
  size_t o = 0;
  auto take = [&](size_t bytes) {
    char* p = ws + o;
    o += (bytes + 255) & ~(size_t)255;
    return p;
  };
  bf16*  Xbf  = (bf16*)take((size_t)NTOK * D_ * 2);   // x in bf16
  bf16*  WinT = (bf16*)take((size_t)N3 * D_ * 2);     // w_in^T bf16 (3072x1024)
  bf16*  WgT  = (bf16*)take((size_t)N2 * D_ * 2);     // w_gate^T bf16 (2048x1024)
  bf16*  Qbf  = (bf16*)take((size_t)NTOK * D_ * 2);   // (B,H,S,HD)
  bf16*  Kbf  = (bf16*)take((size_t)NTOK * D_ * 2);   // (B,H,S,HD)
  bf16*  Vtbf = (bf16*)take((size_t)NTOK * D_ * 2);   // (B,H,HD,S)
  float* bias = (float*)take((size_t)H_ * RLEN * 4);  // TISA rel-pos table
  bf16*  Abf  = (bf16*)take((size_t)NTOK * D_ * 2);   // attn out, token-major
  // total workspace ~50.3 MiB

  cvt_bf16_kernel<<<(NTOK * D_) / 256, 256, 0, stream>>>(x, Xbf, NTOK * D_);
  transpose_bf16_kernel<<<dim3(N3 / 32, D_ / 32), dim3(32, 8), 0, stream>>>(
      w_in, WinT, D_, N3);
  transpose_bf16_kernel<<<dim3(N2 / 32, D_ / 32), dim3(32, 8), 0, stream>>>(
      w_gate, WgT, D_, N2);
  tisa_bias_kernel<<<(H_ * RLEN + 255) / 256, 256, 0, stream>>>(amp, shp, off, bias);
  gemm_qkv_kernel<<<(NTOK / 128) * (N3 / 128), 256, 0, stream>>>(
      Xbf, WinT, Qbf, Kbf, Vtbf);
  attn_kernel<<<B_ * H_ * (S_ / 16 / 8), 256, 0, stream>>>(
      Qbf, Kbf, Vtbf, bias, Abf);
  gate_kernel<<<(NTOK / 128) * (D_ / 64), 256, 0, stream>>>(
      Abf, WgT, b_gate, out);
}
